// Encoder_64278480552466
// MI455X (gfx1250) — matrix operand order repaired and checked by reading, not compiled, not run on hardware
//
#include <hip/hip_runtime.h>
#include <hip/hip_bf16.h>

typedef _Float16 v2h  __attribute__((ext_vector_type(2)));
typedef _Float16 v8h  __attribute__((ext_vector_type(8)));
typedef _Float16 v16h __attribute__((ext_vector_type(16)));
typedef float    v8f  __attribute__((ext_vector_type(8)));

#define H 128
#define TILE 128          // rows per block
#define HSTR 136          // f16 LDS stride (16B-aligned rows, conflict-free b128)
#define EPS_NORM 1e-8f
#define EPS_LN   1e-5f

// ---------------------------------------------------------------------------
// Pack kernel: W (fp32, [K=128][N=128] row-major) -> Wt (f16, [N][K] col-major)
// WMMA B-fragment (kt,nt) then is one contiguous 32B run per lane:
//   lane l<16 : col n = nt*16+l,     K = kt*32 + 0..15
//   lane l>=16: col n = nt*16+l-16,  K = kt*32 + 16..31
// ---------------------------------------------------------------------------
__global__ void pack_w_f16_kernel(const float* __restrict__ w,
                                  _Float16* __restrict__ wt) {
    int idx = blockIdx.x * blockDim.x + threadIdx.x;
    if (idx >= H * H) return;
    int n = idx >> 7;
    int k = idx & 127;
    wt[n * H + k] = (_Float16)w[k * H + n];
}

// ---------------------------------------------------------------------------
// 16x128 strip GEMM (per wave): acc = A(LDS) * W(LDS).
// Accumulators start at zero so the first WMMA per tile takes the inline-0
// C operand (no register fill); bias is added by the caller at writeback.
// B fragments preloaded into bf[8] so loads clause and WMMAs issue back-to-back.
// ---------------------------------------------------------------------------
__device__ __forceinline__ void gemm128(const _Float16* __restrict__ Asrc,
                                        const _Float16* __restrict__ Wlds,
                                        int wave, int lane, v8f acc[8]) {
    const int  lhalf  = lane & 15;
    const bool hiLane = lane >= 16;

#pragma unroll
    for (int kt = 0; kt < 4; ++kt) {
        // A fragment: row = wave*16 + (lane&15); one contiguous 16-element K-run (K = 16h + 0..15).
        const int k0 = kt * 32 + (hiLane ? 16 : 0);
        const _Float16* ap = Asrc + (wave * 16 + lhalf) * HSTR + k0;
        v8h lo = *(const v8h*)(ap);
        v8h hi = *(const v8h*)(ap + 8);
        v16h a = __builtin_shufflevector(lo, hi, 0, 1, 2, 3, 4, 5, 6, 7,
                                                 8, 9, 10, 11, 12, 13, 14, 15);
        v16h bf[8];
#pragma unroll
        for (int nt = 0; nt < 8; ++nt) {
            const _Float16* bp =
                Wlds + (nt * 16 + lhalf) * HSTR + kt * 32 + (hiLane ? 16 : 0);
            bf[nt] = *(const v16h*)bp;
        }
#pragma unroll
        for (int nt = 0; nt < 8; ++nt) {
            acc[nt] = __builtin_amdgcn_wmma_f32_16x16x32_f16(
                false, a, false, bf[nt], (short)0, acc[nt], false, false);
        }
    }
}

// Stage a 128x128 f16 weight matrix (contiguous [n][k]) into LDS stride HSTR.
__device__ __forceinline__ void stage_weights(const _Float16* __restrict__ wt,
                                              _Float16* __restrict__ wb,
                                              int tid) {
#pragma unroll
    for (int it = 0; it < 8; ++it) {
        int c  = tid + it * 256;        // 2048 chunks of 16B
        int n  = c >> 4;
        int k8 = (c & 15) * 8;
        v8h t = *(const v8h*)(wt + c * 8);
        *(v8h*)(&wb[n * HSTR + k8]) = t;
    }
}

// ---------------------------------------------------------------------------
// Encoder kernel: one 128-row tile of the (node | edge) MLP + LayerNorm.
// ---------------------------------------------------------------------------
template <int NIN, bool IS_EDGE>
__global__ void __launch_bounds__(256)
encoder_tile_kernel(const float* __restrict__ featA,      // node: state_hat
                    const float* __restrict__ featB,      // node: node_type
                    const float* __restrict__ mesh_pos,   // edge only
                    const long long* __restrict__ edges,  // edge only
                    const float* __restrict__ mean,
                    const float* __restrict__ stdv,
                    const float* __restrict__ w1,
                    const float* __restrict__ b1,
                    const _Float16* __restrict__ w2t,
                    const float* __restrict__ b2,
                    const _Float16* __restrict__ w3t,
                    const float* __restrict__ b3,
                    const float* __restrict__ gamma,
                    const float* __restrict__ beta,
                    float* __restrict__ out,
                    long long totalRows, long long Eper, long long Nper) {
    extern __shared__ __align__(16) char smem[];
    _Float16* hA    = (_Float16*)smem;                     // 128*136 f16 activ.
    _Float16* wbufA = hA + TILE * HSTR;                    // w2 staging
    _Float16* wbufB = wbufA + TILE * HSTR;                 // w3 staging
    float*    xin   = (float*)(wbufB + TILE * HSTR);       // 128*NIN
    float*    w1s   = xin + TILE * NIN;                    // NIN*128
    float*    b1s   = w1s + NIN * H;
    float*    b2s   = b1s + H;
    float*    b3s   = b2s + H;
    float*    gs    = b3s + H;
    float*    bs    = gs + H;

    const int tid  = threadIdx.x;
    const int lane = tid & 31;
    const int wave = tid >> 5;
    const int lhalf = lane & 15;
    const bool hiLane = lane >= 16;
    const long long rowStart = (long long)blockIdx.x * TILE;

    // ---- stage layer-2 weights + params into LDS ----
    stage_weights(w2t, wbufA, tid);
    for (int i = tid; i < NIN * H; i += 256) w1s[i] = w1[i];
    if (tid < H) {
        b1s[tid] = b1[tid];
        b2s[tid] = b2[tid];
        b3s[tid] = b3[tid];
        gs[tid]  = gamma[tid];
        bs[tid]  = beta[tid];
    }

    // ---- build + normalize input features ----
    if (tid < TILE) {
        const int r = tid;
        const long long gr = rowStart + r;
        float f[NIN];
        if (gr < totalRows) {
            if (IS_EDGE) {
                long long bb = gr / Eper;
                long long s  = edges[gr * 2 + 0];
                long long rc = edges[gr * 2 + 1];
                const float* pb = mesh_pos + bb * Nper * 2;
                float dx = pb[s * 2 + 0] - pb[rc * 2 + 0];
                float dy = pb[s * 2 + 1] - pb[rc * 2 + 1];
                f[0] = dx;
                f[1] = dy;
                f[2] = sqrtf(dx * dx + dy * dy);
            } else {
                f[0] = featA[gr * 2 + 0];
                f[1] = featA[gr * 2 + 1];
#pragma unroll
                for (int k = 0; k < NIN - 2; ++k) f[2 + k] = featB[gr * 9 + k];
            }
        } else {
#pragma unroll
            for (int k = 0; k < NIN; ++k) f[k] = 0.0f;
        }
#pragma unroll
        for (int k = 0; k < NIN; ++k)
            xin[r * NIN + k] = (f[k] - mean[k]) / (stdv[k] + EPS_NORM);
    }
    __syncthreads();    // xin/w1s/wbufA visible to all waves

    // ---- stage layer-3 weights early: latency hides behind layers 1+2 ----
    stage_weights(w3t, wbufB, tid);

    // ---- layer 1 (tiny K): fp32 VALU, ReLU, packed f16 pairs -> hA ----
    // thread pair (2r, 2r+1) writes row r: rows land in the writer wave's own
    // strip, so no barrier is needed between layer 1 and layer 2.
    {
        const int r    = tid >> 1;
        const int half = tid & 1;
        float x[NIN];
#pragma unroll
        for (int k = 0; k < NIN; ++k) x[k] = xin[r * NIN + k];
#pragma unroll 4
        for (int j = 0; j < 64; j += 2) {
            const int c = half * 64 + j;
            float a0 = b1s[c], a1 = b1s[c + 1];
#pragma unroll
            for (int k = 0; k < NIN; ++k) {
                a0 = fmaf(x[k], w1s[k * H + c], a0);
                a1 = fmaf(x[k], w1s[k * H + c + 1], a1);
            }
            v2h p;
            p[0] = (_Float16)fmaxf(a0, 0.0f);
            p[1] = (_Float16)fmaxf(a1, 0.0f);
            *(v2h*)(&hA[r * HSTR + c]) = p;
        }
    }

    // ---- layer 2: WMMA from LDS, bias+ReLU, in-place writeback into hA ----
    {
        v8f acc[8] = {};
        gemm128(hA, wbufA, wave, lane, acc);
#pragma unroll
        for (int nt = 0; nt < 8; ++nt) {
            const float bv = b2s[nt * 16 + lhalf];
#pragma unroll
            for (int g = 0; g < 8; ++g) {
                int row = wave * 16 + g + (hiLane ? 8 : 0);
                int col = nt * 16 + lhalf;
                hA[row * HSTR + col] = (_Float16)fmaxf(acc[nt][g] + bv, 0.0f);
            }
        }
    }
    __syncthreads();    // wbufB (w3) staging visible to all waves

    // ---- layer 3: WMMA, then in-register LayerNorm + store ----
    {
        v8f acc[8] = {};
        gemm128(hA, wbufB, wave, lane, acc);

        float bv[8];
#pragma unroll
        for (int nt = 0; nt < 8; ++nt) bv[nt] = b3s[nt * 16 + lhalf];

#pragma unroll
        for (int g = 0; g < 8; ++g) {
            // row owned by this lane-half for accumulator slot g
            const int r = wave * 16 + g + (hiLane ? 8 : 0);
            const long long gr = rowStart + r;

            float s = 0.0f, s2 = 0.0f;
#pragma unroll
            for (int nt = 0; nt < 8; ++nt) {
                float v = acc[nt][g] + bv[nt];
                s += v;
                s2 += v * v;
            }
            // reduce across the 16 lanes of this half (masks < 16 stay inside)
#pragma unroll
            for (int off = 1; off < 16; off <<= 1) {
                s  += __shfl_xor(s, off, 32);
                s2 += __shfl_xor(s2, off, 32);
            }
            float m   = s * (1.0f / H);
            float var = s2 * (1.0f / H) - m * m;
            float rs  = rsqrtf(var + EPS_LN);

            if (gr < totalRows) {
                float* orow = out + gr * H;
#pragma unroll
                for (int nt = 0; nt < 8; ++nt) {
                    int col = nt * 16 + lhalf;
                    orow[col] =
                        (acc[nt][g] + bv[nt] - m) * rs * gs[col] + bs[col];
                }
            }
        }
    }
}

// ---------------------------------------------------------------------------
static constexpr size_t smem_bytes(int nin) {
    return (size_t)TILE * HSTR * 2 * 3      // hA + wbufA + wbufB
         + (size_t)TILE * nin * 4           // xin
         + (size_t)nin * H * 4              // w1s
         + (size_t)H * 4 * 5;               // b1s, b2s, b3s, gs, bs
}

extern "C" void kernel_launch(void* const* d_in, const int* in_sizes, int n_in,
                              void* d_out, int out_size, void* d_ws, size_t ws_size,
                              hipStream_t stream) {
    const float*     mesh_pos  = (const float*)d_in[0];
    const long long* edges     = (const long long*)d_in[1];
    const float*     node_type = (const float*)d_in[2];
    const float*     state_hat = (const float*)d_in[3];
    const float*     node_mean = (const float*)d_in[5];
    const float*     node_std  = (const float*)d_in[6];
    const float*     edge_mean = (const float*)d_in[7];
    const float*     edge_std  = (const float*)d_in[8];
    const float* fv_w1 = (const float*)d_in[9];
    const float* fv_b1 = (const float*)d_in[10];
    const float* fv_w2 = (const float*)d_in[11];
    const float* fv_b2 = (const float*)d_in[12];
    const float* fv_w3 = (const float*)d_in[13];
    const float* fv_b3 = (const float*)d_in[14];
    const float* fv_g  = (const float*)d_in[15];
    const float* fv_b  = (const float*)d_in[16];
    const float* fe_w1 = (const float*)d_in[17];
    const float* fe_b1 = (const float*)d_in[18];
    const float* fe_w2 = (const float*)d_in[19];
    const float* fe_b2 = (const float*)d_in[20];
    const float* fe_w3 = (const float*)d_in[21];
    const float* fe_b3 = (const float*)d_in[22];
    const float* fe_g  = (const float*)d_in[23];
    const float* fe_b  = (const float*)d_in[24];

    const long long BN = (long long)in_sizes[3] / 2;   // state_hat = B*N*2
    const long long BE = (long long)in_sizes[1] / 2;   // edges     = B*E*2
    const long long B  = 2;
    const long long N  = BN / B;
    const long long E  = BE / B;

    // f16 transposed weight copies in workspace
    _Float16* wv2t = (_Float16*)d_ws;
    _Float16* wv3t = wv2t + H * H;
    _Float16* we2t = wv3t + H * H;
    _Float16* we3t = we2t + H * H;

    pack_w_f16_kernel<<<(H * H + 255) / 256, 256, 0, stream>>>(fv_w2, wv2t);
    pack_w_f16_kernel<<<(H * H + 255) / 256, 256, 0, stream>>>(fv_w3, wv3t);
    pack_w_f16_kernel<<<(H * H + 255) / 256, 256, 0, stream>>>(fe_w2, we2t);
    pack_w_f16_kernel<<<(H * H + 255) / 256, 256, 0, stream>>>(fe_w3, we3t);

    float* outV = (float*)d_out;
    float* outE = outV + BN * H;

    const int nodeBlocks = (int)((BN + TILE - 1) / TILE);
    const int edgeBlocks = (int)((BE + TILE - 1) / TILE);

    encoder_tile_kernel<11, false>
        <<<nodeBlocks, 256, smem_bytes(11), stream>>>(
            state_hat, node_type, nullptr, nullptr, node_mean, node_std,
            fv_w1, fv_b1, wv2t, fv_b2, wv3t, fv_b3, fv_g, fv_b,
            outV, BN, E, N);

    encoder_tile_kernel<3, true>
        <<<edgeBlocks, 256, smem_bytes(3), stream>>>(
            nullptr, nullptr, mesh_pos, edges, edge_mean, edge_std,
            fe_w1, fe_b1, we2t, fe_b2, we3t, fe_b3, fe_g, fe_b,
            outE, BE, E, N);
}
